// StateExpandedBlock_12000138625494
// MI455X (gfx1250) — matrix operand order repaired and checked by reading, not compiled, not run on hardware
//
#include <hip/hip_runtime.h>
#include <math.h>

#define D_MODEL 1024
#define T_LEN   1024
#define N_HEADS 8
#define HEAD_DIM 128
#define LDS_K   48   // padded row stride (elements) to spread LDS banks; 96B, 32B-aligned

typedef __bf16 bf16;
typedef __attribute__((ext_vector_type(8)))  float v8f;
typedef __attribute__((ext_vector_type(8)))  bf16  v8bf;
typedef __attribute__((ext_vector_type(16))) bf16  v16bf;

__device__ __forceinline__ float sigmoidf_(float x) {
    return 1.0f / (1.0f + __expf(-x));
}

// A fragment (contiguous K order): elems 0..15 -> K=base+0..15
__device__ __forceinline__ v16bf load_a_frag(const bf16* p) {
    v8bf a0 = *reinterpret_cast<const v8bf*>(p);
    v8bf a1 = *reinterpret_cast<const v8bf*>(p + 8);
    return __builtin_shufflevector(a0, a1, 0,1,2,3,4,5,6,7,8,9,10,11,12,13,14,15);
}

// bf16x3 accumulate: acc += ahi*bhi + ahi*blo + alo*bhi
__device__ __forceinline__ v8f mm3(v16bf ahi, v16bf alo, v16bf bhi, v16bf blo, v8f acc) {
    acc = __builtin_amdgcn_wmma_f32_16x16x32_bf16(false, ahi, false, bhi, (short)0, acc, false, false);
    acc = __builtin_amdgcn_wmma_f32_16x16x32_bf16(false, ahi, false, blo, (short)0, acc, false, false);
    acc = __builtin_amdgcn_wmma_f32_16x16x32_bf16(false, alo, false, bhi, (short)0, acc, false, false);
    return acc;
}

// Async 16B copy global -> LDS (GVS addressing: SGPR64 base + VGPR u32 byte offset).
// Tracked by ASYNCcnt; LDS address in VDST VGPR.
__device__ __forceinline__ void async_cp16(unsigned lds_off, const bf16* base, unsigned byte_off) {
    asm volatile("global_load_async_to_lds_b128 %0, %1, %2"
                 :: "v"(lds_off), "v"(byte_off), "s"(base) : "memory");
}

// ---------------- Kernel 1: RMSNorm + bf16 hi/lo split ----------------
__global__ __launch_bounds__(256)
void rmsnorm_split_kernel(const float* __restrict__ x, const float* __restrict__ w,
                          bf16* __restrict__ xh, bf16* __restrict__ xl) {
    __shared__ float red[256];
    const int t   = blockIdx.x;
    const int tid = threadIdx.x;
    const float* row = x + (size_t)t * D_MODEL;
    const int base = tid * 4;
    float4 xv = *reinterpret_cast<const float4*>(row + base);
    float v0 = xv.x, v1 = xv.y, v2 = xv.z, v3 = xv.w;
    red[tid] = v0*v0 + v1*v1 + v2*v2 + v3*v3;
    __syncthreads();
    for (int off = 128; off > 0; off >>= 1) {
        if (tid < off) red[tid] += red[tid + off];
        __syncthreads();
    }
    const float scale = rsqrtf(red[0] * (1.0f / D_MODEL) + 1e-6f);
    float vals[4] = {v0, v1, v2, v3};
    for (int k = 0; k < 4; ++k) {
        float xn = vals[k] * scale * w[base + k];
        bf16 hi = (bf16)xn;
        bf16 lo = (bf16)(xn - (float)hi);
        size_t idx = (size_t)t * D_MODEL + base + k;
        xh[idx] = hi;
        xl[idx] = lo;
    }
}

// ---------------- Kernel 2: weight hi/lo split (Wf, Wi, Wo) ----------------
__global__ __launch_bounds__(256)
void weight_split_kernel(const float* __restrict__ Wf, const float* __restrict__ Wi,
                         const float* __restrict__ Wo,
                         bf16* __restrict__ Wh, bf16* __restrict__ Wl) {
    const int n   = blockIdx.x * 256 + threadIdx.x;     // 0 .. 1M-1
    const int mat = blockIdx.y;                          // 0..2
    const float* src = (mat == 0) ? Wf : (mat == 1) ? Wi : Wo;
    float v = src[n];
    bf16 hi = (bf16)v;
    bf16 lo = (bf16)(v - (float)hi);
    size_t idx = (size_t)mat * D_MODEL * D_MODEL + n;
    Wh[idx] = hi;
    Wl[idx] = lo;
}

// ---------------- Kernel 3: bf16x3 WMMA GEMM, async-LDS staged, double-buffered ----
// G = xn @ W^T, fused gate epilogue.
// grid: (16 M-blocks of 64 rows, 8 N-blocks of 128 cols, 3 matrices), 256 threads = 8 waves.
// Block stages A(64x32 hi/lo) + B(128x32 hi/lo) k-chunks in LDS via
// GLOBAL_LOAD_ASYNC_TO_LDS_B128 (ASYNCcnt), double buffered.
// Wave (mq, nq) computes a 32x32 block = 2x2 WMMA tiles, 12 WMMAs / chunk.
__global__ __launch_bounds__(256)
void gemm_bf16x3_kernel(const bf16* __restrict__ xh, const bf16* __restrict__ xl,
                        const bf16* __restrict__ Wh, const bf16* __restrict__ Wl,
                        float* __restrict__ G) {
    __shared__ __attribute__((aligned(64))) bf16 Ah[2][64  * LDS_K];
    __shared__ __attribute__((aligned(64))) bf16 Al[2][64  * LDS_K];
    __shared__ __attribute__((aligned(64))) bf16 Bh[2][128 * LDS_K];
    __shared__ __attribute__((aligned(64))) bf16 Bl[2][128 * LDS_K];

    const int tid  = threadIdx.x;
    const int lane = tid & 31;
    const int wave = tid >> 5;
    const int mat  = blockIdx.z;
    const int t0   = blockIdx.x * 64;
    const int e0b  = blockIdx.y * 128;
    const int hl   = lane >> 4;
    const int lm   = lane & 15;
    const int mq   = wave >> 2;   // 0..1 -> M offset 32*mq
    const int nq   = wave & 3;    // 0..3 -> N offset 32*nq

    const bf16* wh = Wh + (size_t)mat * D_MODEL * D_MODEL;
    const bf16* wl = Wl + (size_t)mat * D_MODEL * D_MODEL;

    // ---- per-thread async copy assignment: row r (0..63), 16B quarter q (0..3)
    const int r = tid >> 2;
    const int q = tid & 3;
    const unsigned gA  = ((unsigned)(t0  + r)      * D_MODEL + q * 8u) * 2u;  // byte off
    const unsigned gB0 = ((unsigned)(e0b + r)      * D_MODEL + q * 8u) * 2u;
    const unsigned gB1 = ((unsigned)(e0b + 64 + r) * D_MODEL + q * 8u) * 2u;
    unsigned ldsA[2], ldsB0[2], ldsB1[2];
    for (int b = 0; b < 2; ++b) {
        ldsA[b]  = (unsigned)(uintptr_t)&Ah[b][r * LDS_K + q * 8];
        ldsB0[b] = (unsigned)(uintptr_t)&Bh[b][r * LDS_K + q * 8];
        ldsB1[b] = (unsigned)(uintptr_t)&Bh[b][(64 + r) * LDS_K + q * 8];
    }
    const unsigned dAl = (unsigned)((uintptr_t)&Al[0][0] - (uintptr_t)&Ah[0][0]);
    const unsigned dBl = (unsigned)((uintptr_t)&Bl[0][0] - (uintptr_t)&Bh[0][0]);

    // ---- per-wave LDS fragment offsets
    const int ar0 = (32 * mq + lm)      * LDS_K + 16 * hl;
    const int ar1 = (32 * mq + 16 + lm) * LDS_K + 16 * hl;
    const int bc0 = (32 * nq + lm)      * LDS_K + 16 * hl;
    const int bc1 = (32 * nq + 16 + lm) * LDS_K + 16 * hl;

    v8f acc00 = {}, acc01 = {}, acc10 = {}, acc11 = {};

    // prologue: stage chunk 0 into buffer 0 (6 async ops per wave)
    {
        const unsigned k0b = 0;
        async_cp16(ldsA[0],        xh, gA  + k0b);
        async_cp16(ldsA[0]  + dAl, xl, gA  + k0b);
        async_cp16(ldsB0[0],       wh, gB0 + k0b);
        async_cp16(ldsB1[0],       wh, gB1 + k0b);
        async_cp16(ldsB0[0] + dBl, wl, gB0 + k0b);
        async_cp16(ldsB1[0] + dBl, wl, gB1 + k0b);
    }

    for (int kc = 0; kc < 32; ++kc) {
        if (kc < 31) {
            const int nb = (kc + 1) & 1;
            const unsigned k0b = (unsigned)(kc + 1) * 64u;  // 32 elems * 2B
            async_cp16(ldsA[nb],        xh, gA  + k0b);
            async_cp16(ldsA[nb]  + dAl, xl, gA  + k0b);
            async_cp16(ldsB0[nb],       wh, gB0 + k0b);
            async_cp16(ldsB1[nb],       wh, gB1 + k0b);
            async_cp16(ldsB0[nb] + dBl, wl, gB0 + k0b);
            async_cp16(ldsB1[nb] + dBl, wl, gB1 + k0b);
            asm volatile("s_wait_asynccnt 0x6" ::: "memory");  // chunk kc landed
        } else {
            asm volatile("s_wait_asynccnt 0x0" ::: "memory");
        }
        __syncthreads();

        const int buf = kc & 1;
        v16bf ahi0 = load_a_frag(&Ah[buf][ar0]);
        v16bf alo0 = load_a_frag(&Al[buf][ar0]);
        v16bf ahi1 = load_a_frag(&Ah[buf][ar1]);
        v16bf alo1 = load_a_frag(&Al[buf][ar1]);
        v16bf bhi0 = *reinterpret_cast<const v16bf*>(&Bh[buf][bc0]);
        v16bf blo0 = *reinterpret_cast<const v16bf*>(&Bl[buf][bc0]);
        v16bf bhi1 = *reinterpret_cast<const v16bf*>(&Bh[buf][bc1]);
        v16bf blo1 = *reinterpret_cast<const v16bf*>(&Bl[buf][bc1]);
        acc00 = mm3(ahi0, alo0, bhi0, blo0, acc00);
        acc01 = mm3(ahi0, alo0, bhi1, blo1, acc01);
        acc10 = mm3(ahi1, alo1, bhi0, blo0, acc10);
        acc11 = mm3(ahi1, alo1, bhi1, blo1, acc11);

        __syncthreads();  // all waves done reading buf before it is re-filled
    }

    // Epilogue: C/D layout -> VGPR rr, lane l: M = rr + 8*(l>=16), N = l%16
    float* Gm = G + (size_t)mat * D_MODEL * T_LEN;
    const v8f* accs[4] = {&acc00, &acc01, &acc10, &acc11};
    for (int qd = 0; qd < 4; ++qd) {
        const int mt = qd >> 1, nt = qd & 1;
        const v8f& a = *accs[qd];
        for (int rr = 0; rr < 8; ++rr) {
            int row = t0 + 32 * mq + mt * 16 + rr + 8 * hl;
            int col = e0b + 32 * nq + nt * 16 + lm;
            float val = a[rr];
            if (mat == 0)      val = fmaxf(sigmoidf_(val), 0.9f);  // forget gate
            else if (mat == 2) val = sigmoidf_(val);               // output gate
            Gm[(size_t)row * D_MODEL + col] = val;
        }
    }
}

// ---------------- Kernel 4: per-(t,head) sum of v ----------------
__global__ __launch_bounds__(256)
void vsum_kernel(const float* __restrict__ V, float* __restrict__ Vs) {
    const int g = blockIdx.x * 256 + threadIdx.x;  // 0..8191
    const int t = g >> 3, h = g & 7;
    const float* p = V + (size_t)t * D_MODEL + h * HEAD_DIM;
    float s = 0.0f;
    for (int j = 0; j < HEAD_DIM; ++j) s += p[j];
    Vs[t * N_HEADS + h] = s;
}

// ---------------- Kernels 5-7: chunked scan over T (32 chunks of 32) ----------------
__global__ __launch_bounds__(256)
void scan_part1_kernel(const float* __restrict__ F, const float* __restrict__ Vs,
                       float* __restrict__ Pc, float* __restrict__ Sc) {
    const int g  = blockIdx.x * 256 + threadIdx.x;  // 0..32767
    const int c  = g >> 10;
    const int ch = g & 1023;
    const int h  = ch >> 7;
    float P = 1.0f, S = 0.0f;
    for (int j = 0; j < 32; ++j) {
        int t = c * 32 + j;
        float f  = F[(size_t)t * D_MODEL + ch];
        float fd = f + 1e-8f;
        S = fd * S + (1.0f - f) * Vs[t * N_HEADS + h];
        P *= fd;
    }
    Pc[c * D_MODEL + ch] = P;
    Sc[c * D_MODEL + ch] = S;
}

__global__ __launch_bounds__(256)
void scan_carry_kernel(const float* __restrict__ Pc, const float* __restrict__ Sc,
                       float* __restrict__ Kc) {
    const int ch = blockIdx.x * 256 + threadIdx.x;  // 0..1023
    float carry = 0.0f;
    for (int c = 0; c < 32; ++c) {
        Kc[c * D_MODEL + ch] = carry;
        carry = Pc[c * D_MODEL + ch] * carry + Sc[c * D_MODEL + ch];
    }
}

__global__ __launch_bounds__(256)
void scan_part2_kernel(const float* __restrict__ F, const float* __restrict__ Vs,
                       const float* __restrict__ Kc, const float* __restrict__ x,
                       const float* __restrict__ O, float* __restrict__ out) {
    const int g  = blockIdx.x * 256 + threadIdx.x;
    const int c  = g >> 10;
    const int ch = g & 1023;
    const int h  = ch >> 7;
    float cst = Kc[c * D_MODEL + ch];
    for (int j = 0; j < 32; ++j) {
        int t = c * 32 + j;
        size_t idx = (size_t)t * D_MODEL + ch;
        float f  = F[idx];
        float fd = f + 1e-8f;
        cst = fd * cst + (1.0f - f) * Vs[t * N_HEADS + h];
        out[idx] = x[idx] + O[idx] * cst;
    }
}

extern "C" void kernel_launch(void* const* d_in, const int* in_sizes, int n_in,
                              void* d_out, int out_size, void* d_ws, size_t ws_size,
                              hipStream_t stream) {
    (void)in_sizes; (void)n_in; (void)out_size; (void)ws_size;
    const float* x  = (const float*)d_in[0];
    const float* nw = (const float*)d_in[1];
    const float* Wf = (const float*)d_in[2];
    const float* Wi = (const float*)d_in[3];
    const float* Wo = (const float*)d_in[4];
    float* out = (float*)d_out;

    char* ws = (char*)d_ws;
    const size_t MB = 1u << 20;
    bf16*  xh = (bf16*) (ws);                       // 2 MB
    bf16*  xl = (bf16*) (ws + 2 * MB);              // 2 MB
    bf16*  Wh = (bf16*) (ws + 4 * MB);              // 6 MB (3 matrices)
    bf16*  Wl = (bf16*) (ws + 10 * MB);             // 6 MB
    float* G  = (float*)(ws + 16 * MB);             // 12 MB: [F | V | O]
    float* Vs = (float*)(ws + 28 * MB);             // 32 KB
    float* Pc = (float*)(ws + 28 * MB + 128 * 1024);
    float* Sc = (float*)(ws + 28 * MB + 256 * 1024);
    float* Kc = (float*)(ws + 28 * MB + 384 * 1024);

    rmsnorm_split_kernel<<<T_LEN, 256, 0, stream>>>(x, nw, xh, xl);
    weight_split_kernel<<<dim3(4096, 3), 256, 0, stream>>>(Wf, Wi, Wo, Wh, Wl);
    gemm_bf16x3_kernel<<<dim3(16, 8, 3), 256, 0, stream>>>(xh, xl, Wh, Wl, G);
    vsum_kernel<<<32, 256, 0, stream>>>(G + (size_t)1 * D_MODEL * T_LEN, Vs);
    scan_part1_kernel<<<128, 256, 0, stream>>>(G, Vs, Pc, Sc);
    scan_carry_kernel<<<4, 256, 0, stream>>>(Pc, Sc, Kc);
    scan_part2_kernel<<<128, 256, 0, stream>>>(G, Vs, Kc, x,
                                               G + (size_t)2 * D_MODEL * T_LEN, out);
}
